// Attention_73220602462356
// MI455X (gfx1250) — matrix operand order repaired and checked by reading, not compiled, not run on hardware
//
#include <hip/hip_runtime.h>
#include <hip/hip_bf16.h>

#define DIM      1024
#define N_HEAD   16
#define HEAD_DIM 64
#define BATCH    4
#define SEQ      2048
#define MTOT     (BATCH * SEQ)      // 8192 rows
#define LDSROW   40                 // padded LDS row stride (halves) to dodge bank conflicts

typedef __attribute__((ext_vector_type(16))) __bf16 bf16x16;
typedef __attribute__((ext_vector_type(8)))  float  f32x8;

union FragU { uint4 q[2]; bf16x16 v; unsigned short h[16]; };

__device__ __forceinline__ unsigned short f32_to_bf16(float f) {
    union { float f; unsigned int u; } x; x.f = f;
    unsigned int r = x.u + 0x7FFFu + ((x.u >> 16) & 1u);   // round-to-nearest-even
    return (unsigned short)(r >> 16);
}

__device__ __forceinline__ f32x8 wmma_bf16(bf16x16 a, bf16x16 b, f32x8 c) {
    // D = A(16x32) x B(32x16) + C, f32 accumulate
    return __builtin_amdgcn_wmma_f32_16x16x32_bf16(false, a, false, b, (short)0, c,
                                                   false, false);
}

// A fragment (16x32, row-major, K contiguous, leading dim ld halves).
// ISA layout: lane row = lane&15; lanes<16 K={0..7,16..23}, lanes>=16 K={8..15,24..31}.
__device__ __forceinline__ bf16x16 load_frag_a(const unsigned short* p, int ld) {
    int lane = threadIdx.x & 31;
    const unsigned short* q = p + (lane & 15) * ld + ((lane >> 4) << 3);
    FragU f;
    f.q[0] = *(const uint4*)(q);
    f.q[1] = *(const uint4*)(q + 16);
    return f.v;
}

// B fragment (32x16) with COLUMNS contiguous in memory (column base stride ld halves).
// ISA layout: lane col = lane&15; lanes<16 K={0..7,16..23}, lanes>=16 K={8..15,24..31} (packed 2/VGPR).
__device__ __forceinline__ bf16x16 load_frag_b_colmajor(const unsigned short* p, int ld) {
    int lane = threadIdx.x & 31;
    const unsigned short* q = p + (lane & 15) * ld + ((lane >> 4) << 3);
    FragU f;
    f.q[0] = *(const uint4*)(q);
    f.q[1] = *(const uint4*)(q + 16);
    return f.v;
}

// ---------------------------------------------------------------- convert f32 -> bf16
__global__ void cvt_bf16(const float* __restrict__ in, unsigned short* __restrict__ out, int n) {
    int i = blockIdx.x * blockDim.x + threadIdx.x;
    if (i < n) out[i] = f32_to_bf16(in[i]);
}

// ---------------------------------------------------------------- C[M,N] = A[M,K] * W[N,K]^T
// Per-wave 32x64 tile: 8 x v_wmma_f32_16x16x32_bf16 per 32-wide K step.
__global__ __launch_bounds__(256) void gemm_xwT(const unsigned short* __restrict__ A,
                                                const unsigned short* __restrict__ W,
                                                unsigned short* __restrict__ outb,
                                                float* __restrict__ outf,
                                                int M, int N, int K) {
    int gw   = (blockIdx.x * blockDim.x + threadIdx.x) >> 5;
    int mtiles = M / 32;
    int mt = gw % mtiles;
    int nt = gw / mtiles;
    int lane = threadIdx.x & 31;

    f32x8 acc[2][4] = {};
    const unsigned short* Ab = A + (size_t)(mt * 32) * K;
    const unsigned short* Wb = W + (size_t)(nt * 64) * K;

    for (int k0 = 0; k0 < K; k0 += 32) {
        bf16x16 a0 = load_frag_a(Ab + k0, K);
        bf16x16 a1 = load_frag_a(Ab + 16 * K + k0, K);
#pragma unroll
        for (int j = 0; j < 4; ++j) {
            bf16x16 b = load_frag_b_colmajor(Wb + (j * 16) * K + k0, K);
            acc[0][j] = wmma_bf16(a0, b, acc[0][j]);
            acc[1][j] = wmma_bf16(a1, b, acc[1][j]);
        }
    }
    // C layout: VGPR i -> row i (lanes<16) / i+8 (lanes>=16), col = lane&15
    int nloc = lane & 15;
    int moff = (lane >> 4) * 8;
#pragma unroll
    for (int i2 = 0; i2 < 2; ++i2)
#pragma unroll
        for (int j = 0; j < 4; ++j)
#pragma unroll
            for (int i = 0; i < 8; ++i) {
                size_t row = mt * 32 + i2 * 16 + moff + i;
                size_t col = nt * 64 + j * 16 + nloc;
                float vv = acc[i2][j][i];
                if (outf) outf[row * N + col] = vv;
                else      outb[row * N + col] = f32_to_bf16(vv);
            }
}

// ---------------------------------------------------------------- V transpose
// vin [b*SEQ + t][d] -> vt [b][d][t], LDS-tiled 32x32 so both sides stay coalesced.
__global__ __launch_bounds__(256) void transpose_v(const unsigned short* __restrict__ vin,
                                                   unsigned short* __restrict__ vt) {
    __shared__ unsigned short tile[32][34];   // +2 pad: kills bank conflicts on the gather
    int blk = blockIdx.x;
    int dd = blk & 31;                 // DIM/32 tiles
    int tt = (blk >> 5) & 63;          // SEQ/32 tiles
    int b  = blk >> 11;
    int tid = threadIdx.x;
    int r  = tid >> 3;                 // 0..31
    int c4 = (tid & 7) * 4;            // 0,4,...,28

    const unsigned short* src =
        vin + ((size_t)(b * SEQ + tt * 32 + r)) * DIM + dd * 32 + c4;
    uint2 ld = *(const uint2*)src;     // 4 halves, coalesced b64
    const unsigned short* lh = (const unsigned short*)&ld;
#pragma unroll
    for (int i = 0; i < 4; ++i) tile[r][c4 + i] = lh[i];
    __syncthreads();

    size_t dglob = (size_t)dd * 32 + r;
    unsigned short* dst = vt + ((size_t)b * DIM + dglob) * SEQ + tt * 32 + c4;
#pragma unroll
    for (int i = 0; i < 4; ++i) dst[i] = tile[c4 + i][r];
}

// ---------------------------------------------------------------- flash attention
// One wave = one (batch, head, 16-query tile). Online softmax over 32-key chunks.
// K rows and V^T columns are both contiguous -> all fragments are b128 loads.
__global__ __launch_bounds__(128) void attn_kernel(const unsigned short* __restrict__ q,
                                                   const unsigned short* __restrict__ k,
                                                   const unsigned short* __restrict__ vt,
                                                   unsigned short* __restrict__ attn_out) {
    __shared__ unsigned short plds[4 * 16 * LDSROW];
    int wid  = threadIdx.x >> 5;
    int lane = threadIdx.x & 31;
    int gw   = blockIdx.x * 4 + wid;
    int qt = gw & 127;                 // SEQ/16
    int h  = (gw >> 7) & (N_HEAD - 1);
    int b  = gw >> 11;

    int qg0  = qt * 16;
    int nloc = lane & 15;
    int moff = (lane >> 4) * 8;
    unsigned short* myLds = plds + wid * 16 * LDSROW;

    const unsigned short* Qb = q + ((size_t)(b * SEQ + qg0)) * DIM + h * HEAD_DIM;
    const unsigned short* Vtb = vt + ((size_t)(b * DIM + h * HEAD_DIM)) * SEQ;
    bf16x16 qa0 = load_frag_a(Qb + 0,  DIM);
    bf16x16 qa1 = load_frag_a(Qb + 32, DIM);

    f32x8 o[4] = {};
    float mrow[8], lrow[8];
#pragma unroll
    for (int i = 0; i < 8; ++i) { mrow[i] = -1e30f; lrow[i] = 0.f; }
    const float scale = 0.125f;   // 1/sqrt(64)

    for (int kc = 0; kc < qg0 + 16; kc += 32) {
        // prefetch next chunk's K rows and V^T columns (global_prefetch_b8, 1B/lane
        // pulls whole cachelines; speculative, silently dropped if useless)
        if (kc + 32 < qg0 + 16) {
            __builtin_prefetch(k + ((size_t)(b * SEQ + kc + 32 + lane)) * DIM + h * HEAD_DIM, 0, 3);
            __builtin_prefetch(Vtb + ((size_t)(lane)      * SEQ) + kc + 32, 0, 3);
            __builtin_prefetch(Vtb + ((size_t)(lane + 32) * SEQ) + kc + 32, 0, 3);
        }
        // ---- S = scale * Q K^T (two 16-key N tiles), causal mask applied analytically
        f32x8 s[2];
#pragma unroll
        for (int j = 0; j < 2; ++j) {
            const unsigned short* Kb = k + ((size_t)(b * SEQ + kc + j * 16)) * DIM + h * HEAD_DIM;
            f32x8 sa = {};
            sa = wmma_bf16(qa0, load_frag_b_colmajor(Kb + 0,  DIM), sa);
            sa = wmma_bf16(qa1, load_frag_b_colmajor(Kb + 32, DIM), sa);
            int keyg = kc + j * 16 + nloc;
#pragma unroll
            for (int i = 0; i < 8; ++i) {
                int qrow = qg0 + moff + i;
                s[j][i] = (keyg > qrow) ? -1e9f : sa[i] * scale;
            }
        }
        // ---- online softmax: row stats live per-lane (rows moff..moff+7), reduce width 16
        float mnew[8], corr[8];
#pragma unroll
        for (int i = 0; i < 8; ++i) {
            float mx = fmaxf(s[0][i], s[1][i]);
            mx = fmaxf(mx, __shfl_xor(mx, 1, 16));
            mx = fmaxf(mx, __shfl_xor(mx, 2, 16));
            mx = fmaxf(mx, __shfl_xor(mx, 4, 16));
            mx = fmaxf(mx, __shfl_xor(mx, 8, 16));
            mnew[i] = fmaxf(mrow[i], mx);
            corr[i] = __expf(mrow[i] - mnew[i]);
            mrow[i] = mnew[i];
        }
#pragma unroll
        for (int j = 0; j < 2; ++j)
#pragma unroll
            for (int i = 0; i < 8; ++i)
                s[j][i] = __expf(s[j][i] - mnew[i]);
#pragma unroll
        for (int i = 0; i < 8; ++i) {
            float r = s[0][i] + s[1][i];
            r += __shfl_xor(r, 1, 16);
            r += __shfl_xor(r, 2, 16);
            r += __shfl_xor(r, 4, 16);
            r += __shfl_xor(r, 8, 16);
            lrow[i] = lrow[i] * corr[i] + r;
        }
#pragma unroll
        for (int jt = 0; jt < 4; ++jt)
#pragma unroll
            for (int i = 0; i < 8; ++i)
                o[jt][i] *= corr[i];

        // ---- restage P (C layout) -> A fragment via LDS
#pragma unroll
        for (int j = 0; j < 2; ++j)
#pragma unroll
            for (int i = 0; i < 8; ++i)
                myLds[(moff + i) * LDSROW + j * 16 + nloc] = f32_to_bf16(s[j][i]);
        asm volatile("s_wait_dscnt 0" ::: "memory");   // same-wave RAW through LDS
        FragU pf;
        {
            const unsigned short* base = myLds + (lane & 15) * LDSROW + ((lane >> 4) << 3);
            pf.q[0] = *(const uint4*)(base);
            pf.q[1] = *(const uint4*)(base + 16);
        }
        // ---- O += P(16x32) * V(32x64); V^T columns are contiguous -> b128 fragments
#pragma unroll
        for (int jt = 0; jt < 4; ++jt) {
            const unsigned short* Vb = Vtb + ((size_t)(jt * 16)) * SEQ + kc;
            bf16x16 vb = load_frag_b_colmajor(Vb, SEQ);
            o[jt] = wmma_bf16(pf.v, vb, o[jt]);
        }
    }
    // ---- epilogue: normalize by l, write bf16 [MTOT, DIM]
#pragma unroll
    for (int i = 0; i < 8; ++i) lrow[i] = 1.f / lrow[i];
#pragma unroll
    for (int jt = 0; jt < 4; ++jt)
#pragma unroll
        for (int i = 0; i < 8; ++i) {
            size_t row = (size_t)(b * SEQ + qg0 + moff + i);
            attn_out[row * DIM + h * HEAD_DIM + jt * 16 + nloc] =
                f32_to_bf16(o[jt][i] * lrow[i]);
        }
}

extern "C" void kernel_launch(void* const* d_in, const int* in_sizes, int n_in,
                              void* d_out, int out_size, void* d_ws, size_t ws_size,
                              hipStream_t stream) {
    const float* x  = (const float*)d_in[0];
    // d_in[1] is the causal mask; applied analytically inside attn_kernel (identical values)
    const float* wq = (const float*)d_in[2];
    const float* wk = (const float*)d_in[3];
    const float* wv = (const float*)d_in[4];
    const float* wo = (const float*)d_in[5];
    float* out = (float*)d_out;

    unsigned short* xb  = (unsigned short*)d_ws;
    unsigned short* wqb = xb  + (size_t)MTOT * DIM;
    unsigned short* wkb = wqb + (size_t)DIM * DIM;
    unsigned short* wvb = wkb + (size_t)DIM * DIM;
    unsigned short* wob = wvb + (size_t)DIM * DIM;
    unsigned short* qb  = wob + (size_t)DIM * DIM;
    unsigned short* kb  = qb  + (size_t)MTOT * DIM;
    unsigned short* vb  = kb  + (size_t)MTOT * DIM;
    unsigned short* vtb = vb  + (size_t)MTOT * DIM;
    unsigned short* ab  = vtb + (size_t)MTOT * DIM;

    const int nx = MTOT * DIM;          // 8388608
    const int nw = DIM * DIM;           // 1048576
    cvt_bf16<<<(nx + 255) / 256, 256, 0, stream>>>(x,  xb,  nx);
    cvt_bf16<<<(nw + 255) / 256, 256, 0, stream>>>(wq, wqb, nw);
    cvt_bf16<<<(nw + 255) / 256, 256, 0, stream>>>(wk, wkb, nw);
    cvt_bf16<<<(nw + 255) / 256, 256, 0, stream>>>(wv, wvb, nw);
    cvt_bf16<<<(nw + 255) / 256, 256, 0, stream>>>(wo, wob, nw);

    // (MTOT/32) * (DIM/64) = 4096 waves, 8 waves/block -> 512 blocks
    const int gblocks = (MTOT / 32) * (DIM / 64) / 8;
    gemm_xwT<<<gblocks, 256, 0, stream>>>(xb, wqb, qb, nullptr, MTOT, DIM, DIM);
    gemm_xwT<<<gblocks, 256, 0, stream>>>(xb, wkb, kb, nullptr, MTOT, DIM, DIM);
    gemm_xwT<<<gblocks, 256, 0, stream>>>(xb, wvb, vb, nullptr, MTOT, DIM, DIM);

    // BATCH * (SEQ/32) * (DIM/32) = 8192 tiles
    transpose_v<<<8192, 256, 0, stream>>>(vb, vtb);

    // BATCH * N_HEAD * (SEQ/16) = 8192 waves, 4 waves/block -> 2048 blocks
    attn_kernel<<<2048, 128, 0, stream>>>(qb, kb, vtb, ab);

    gemm_xwT<<<gblocks, 256, 0, stream>>>(ab, wob, nullptr, out, MTOT, DIM, DIM);
}
